// MultistageSpatialContext_62354335203889
// MI455X (gfx1250) — matrix operand order repaired and checked by reading, not compiled, not run on hardware
//
#include <hip/hip_runtime.h>

// ---- vector types ----
typedef __attribute__((ext_vector_type(16))) __bf16          v16bf;
typedef __attribute__((ext_vector_type(8)))  float           v8f;
typedef __attribute__((ext_vector_type(8)))  unsigned short  u16x8;
typedef __attribute__((ext_vector_type(16))) unsigned short  u16x16;
typedef __attribute__((ext_vector_type(4)))  unsigned int    u32x4;
typedef __attribute__((ext_vector_type(8)))  int             i32x8;
typedef __attribute__((ext_vector_type(4)))  int             i32x4;

// ---- problem constants ----
#define BATCH  8
#define CIN    192
#define COUT   384
#define HH     128
#define WW     128
#define NTAPS  12              // taps with dh+dw odd (only survivors at step=2)
#define CCHUNK 32              // WMMA K per step
#define NCC    (CIN / CCHUNK)  // 6
#define NKG    (NTAPS * NCC)   // 72
#define NCOG   (COUT / 16)     // 24

// padded pre-converted y image (halo: 2 rows top/bottom, 1 left / 3 right -> 132x132)
#define PH 132
#define PW 132

// LDS staged tile: 11 rows x 35 cols x 32 ch (bf16), contiguous as TDM writes it
#define LROWS 11
#define LCOLS 35
#define TILE_SHORTS (LROWS * LCOLS * CCHUNK)   // 12320 shorts = 24,640 B

static __device__ __forceinline__ unsigned short f2bf(float f) {
  unsigned int u = __float_as_uint(f);
  unsigned int r = u + 0x7fffu + ((u >> 16) & 1u);   // round-to-nearest-even
  return (unsigned short)(r >> 16);
}

// -------------------------------------------------------------------------
// Pre-pass 1: pack weight[384][192][5][5] fp32 -> bf16 B-fragments.
// bp[((kg*NCOG + cog)*32 + lane)*16 + e], kg = tap*NCC+cc,
// B 32x16 layout: N = lane&15, K = 16*(lane>=16) + e
// -------------------------------------------------------------------------
__global__ __launch_bounds__(256) void pack_weights_bf16(
    const float* __restrict__ w, unsigned short* __restrict__ bp) {
  constexpr int TDH[NTAPS] = {0,0,1,1,1,2,2,3,3,3,4,4};
  constexpr int TDW[NTAPS] = {1,3,0,2,4,1,3,0,2,4,1,3};
  int idx = blockIdx.x * blockDim.x + threadIdx.x;
  constexpr int TOTAL = NKG * NCOG * 32 * 16;
  if (idx >= TOTAL) return;
  int e    = idx & 15;
  int lane = (idx >> 4) & 31;
  int grp  = idx >> 9;
  int cog  = grp % NCOG;
  int kg   = grp / NCOG;
  int tap  = kg / NCC;
  int cc   = kg - tap * NCC;
  int co   = cog * 16 + (lane & 15);
  int ci   = cc * CCHUNK + ((lane >> 4) << 4) + e;
  bp[idx]  = f2bf(w[((size_t)co * CIN + ci) * 25 + TDH[tap] * 5 + TDW[tap]]);
}

// -------------------------------------------------------------------------
// Pre-pass 2: y fp32 NCHW -> padded bf16 [b][cc][r(132)][c(132)][ci(32)].
// Halo zero-filled so TDM tiles never need bounds checks.
// -------------------------------------------------------------------------
__global__ __launch_bounds__(256) void pack_y_bf16(
    const float* __restrict__ y, unsigned short* __restrict__ ybf) {
  size_t idx = (size_t)blockIdx.x * blockDim.x + threadIdx.x;
  const size_t TOTAL = (size_t)BATCH * NCC * PH * PW * CCHUNK;
  if (idx >= TOTAL) return;
  int ci = (int)(idx & 31);
  size_t p = idx >> 5;
  int c  = (int)(p % PW); p /= PW;
  int r  = (int)(p % PH); p /= PH;
  int cc = (int)(p % NCC);
  int b  = (int)(p / NCC);
  int ih = r - 2;
  int iw = c - 1;
  float v = 0.f;
  if ((unsigned)ih < (unsigned)HH && (unsigned)iw < (unsigned)WW)
    v = y[(((size_t)b * CIN + cc * CCHUNK + ci) * HH + ih) * WW + iw];
  ybf[idx] = f2bf(v);
}

// -------------------------------------------------------------------------
// TDM: one 2D tile (11 rows x 2240 B, row pitch 8448 B) global -> LDS.
// data_size = 4B units: tile_dim0 = 560, tile_dim1 = 11, stride0 = 2112.
// Tensor dims set huge (bounds handled by the padded image).
// -------------------------------------------------------------------------
static __device__ __forceinline__ void tdm_load_tile(
    unsigned long long gaddr, unsigned int lds_addr) {
  u32x4 g0;
  g0[0] = 1u;                                          // count = 1 valid descriptor
  g0[1] = lds_addr;                                    // LDS byte address
  g0[2] = (unsigned int)(gaddr & 0xffffffffull);       // global_addr[31:0]
  g0[3] = (unsigned int)((gaddr >> 32) & 0x01ffffffull) | (2u << 30);  // [56:32] | type=2
  i32x8 g1;
  g1[0] = (int)(2u << 16);                             // data_size = 4B
  g1[1] = 0;                                           // tensor_dim0[15:0] (=0, hi below)
  g1[2] = 0x0400;                                      // tensor_dim0 = 1<<26 ; tdim1 lo = 0
  g1[3] = (560 << 16) | 0x0400;                        // tile_dim0 = 560 ; tensor_dim1 = 1<<26
  g1[4] = 11;                                          // tile_dim1 = 11, tile_dim2 = 0 (2D)
  g1[5] = 2112;                                        // tensor_dim0_stride (dwords)
  g1[6] = 0;
  g1[7] = 0;
  i32x4 gz4 = {0, 0, 0, 0};
#if defined(__clang_major__) && (__clang_major__ >= 23)
  i32x8 gz8 = {0, 0, 0, 0, 0, 0, 0, 0};
  __builtin_amdgcn_tensor_load_to_lds(g0, g1, gz4, gz4, gz8, 0);
#else
  __builtin_amdgcn_tensor_load_to_lds(g0, g1, gz4, gz4, 0);
#endif
}

// -------------------------------------------------------------------------
// Main kernel: implicit GEMM over active pixels (h even, w odd).
// Block: 64 active pixels x 128 Cout; 8 waves = 4(M) x 2(N).
// TDM double-buffers the y tile; B fragments software-pipelined one tap ahead.
// -------------------------------------------------------------------------
__global__ __launch_bounds__(256) void msc_step2_wmma(
    const unsigned short* __restrict__ ybf,
    const unsigned short* __restrict__ bp,
    float* __restrict__ out) {
  constexpr int TDH[NTAPS] = {0,0,1,1,1,2,2,3,3,3,4,4};
  constexpr int TDW[NTAPS] = {1,3,0,2,4,1,3,0,2,4,1,3};

  __shared__ unsigned short yt[2][TILE_SHORTS];        // 49,280 B double buffer

  int nblk = blockIdx.x % 3;
  int mblk = blockIdx.x / 3;
  int b    = mblk >> 6;
  int rb   = (mblk & 63) >> 2;
  int cb   = mblk & 3;
  int h0   = rb * 8;
  int w0   = cb * 32;

  int tid    = threadIdx.x;
  int lane   = tid & 31;
  int wave   = tid >> 5;
  int waveM  = wave & 3;
  int waveN  = wave >> 2;
  int lm     = lane & 15;
  int laneHi = lane >> 4;
  int cog0   = nblk * 8 + waveN * 4;

  // per-cc tile start in padded bf16 image: row = h0, col = w0
  const size_t ccStrideB = (size_t)PH * PW * CCHUNK * 2;           // bytes per cc image
  unsigned long long gbase = (unsigned long long)(uintptr_t)ybf +
      ((((size_t)b * NCC) * PH + h0) * PW + w0) * CCHUNK * 2;

  v8f acc[4] = {v8f{}, v8f{}, v8f{}, v8f{}};

  // prologue: DMA chunk 0 into buffer 0
  if (wave == 0) {
    tdm_load_tile(gbase, (unsigned int)(uintptr_t)&yt[0][0]);
    __builtin_amdgcn_s_wait_tensorcnt(0);
  }
  __syncthreads();

  for (int cc = 0; cc < NCC; ++cc) {
    int buf = cc & 1;
    if (wave == 0 && cc + 1 < NCC) {                   // overlap next DMA with compute
      tdm_load_tile(gbase + (size_t)(cc + 1) * ccStrideB,
                    (unsigned int)(uintptr_t)&yt[buf ^ 1][0]);
    }
    const unsigned short* tb = &yt[buf][0];

    // ---- B fragment pipeline: prime tap 0 ----
    const u16x16* bpt0 = reinterpret_cast<const u16x16*>(
        bp + (((size_t)(0 * NCC + cc) * NCOG + cog0) * 32 + lane) * 16);
    u16x16 bc0 = bpt0[0], bc1 = bpt0[32], bc2 = bpt0[64], bc3 = bpt0[96];

#pragma unroll
    for (int tap = 0; tap < NTAPS; ++tap) {
      // prefetch-load B for tap+1 into a second live set
      u16x16 bn0, bn1, bn2, bn3;
      if (tap + 1 < NTAPS) {
        const u16x16* bpn = reinterpret_cast<const u16x16*>(
            bp + (((size_t)((tap + 1) * NCC + cc) * NCOG + cog0) * 32 + lane) * 16);
        bn0 = bpn[0]; bn1 = bpn[32]; bn2 = bpn[64]; bn3 = bpn[96];
      }

      // A fragment from LDS: M = lane&15 (active col), K = 16*laneHi + {0..15}
      int row  = 2 * waveM + TDH[tap];
      int col  = 2 * lm + TDW[tap];
      int base = (row * LCOLS + col) * CCHUNK + laneHi * 16;
      u16x8 alo = *reinterpret_cast<const u16x8*>(&tb[base]);
      u16x8 ahi = *reinterpret_cast<const u16x8*>(&tb[base + 8]);
      u16x16 av = __builtin_shufflevector(alo, ahi,
          0,1,2,3,4,5,6,7,8,9,10,11,12,13,14,15);
      v16bf a = __builtin_bit_cast(v16bf, av);

      acc[0] = __builtin_amdgcn_wmma_f32_16x16x32_bf16(
          false, a, false, __builtin_bit_cast(v16bf, bc0), (short)0, acc[0], false, false);
      acc[1] = __builtin_amdgcn_wmma_f32_16x16x32_bf16(
          false, a, false, __builtin_bit_cast(v16bf, bc1), (short)0, acc[1], false, false);
      acc[2] = __builtin_amdgcn_wmma_f32_16x16x32_bf16(
          false, a, false, __builtin_bit_cast(v16bf, bc2), (short)0, acc[2], false, false);
      acc[3] = __builtin_amdgcn_wmma_f32_16x16x32_bf16(
          false, a, false, __builtin_bit_cast(v16bf, bc3), (short)0, acc[3], false, false);

      bc0 = bn0; bc1 = bn1; bc2 = bn2; bc3 = bn3;
    }

    if (wave == 0 && cc + 1 < NCC)
      __builtin_amdgcn_s_wait_tensorcnt(0);            // publish next buffer
    __syncthreads();
  }

  // ---- store: interleave the three zero pixels of each 2x2 decode patch ----
  int hA    = h0 + 2 * waveM;
  int wbase = w0 + 16 * laneHi;
#pragma unroll
  for (int f = 0; f < 4; ++f) {
    int co = (cog0 + f) * 16 + lm;
    float* po = out + (((size_t)b * COUT + co) * HH + hA) * WW + wbase;
    v8f A = acc[f];
    *reinterpret_cast<float4*>(po +  0) = make_float4(0.f, A[0], 0.f, A[1]);
    *reinterpret_cast<float4*>(po +  4) = make_float4(0.f, A[2], 0.f, A[3]);
    *reinterpret_cast<float4*>(po +  8) = make_float4(0.f, A[4], 0.f, A[5]);
    *reinterpret_cast<float4*>(po + 12) = make_float4(0.f, A[6], 0.f, A[7]);
    float* pz = po + WW;
    float4 z = make_float4(0.f, 0.f, 0.f, 0.f);
    *reinterpret_cast<float4*>(pz +  0) = z;
    *reinterpret_cast<float4*>(pz +  4) = z;
    *reinterpret_cast<float4*>(pz +  8) = z;
    *reinterpret_cast<float4*>(pz + 12) = z;
  }
}

// -------------------------------------------------------------------------
extern "C" void kernel_launch(void* const* d_in, const int* in_sizes, int n_in,
                              void* d_out, int out_size, void* d_ws, size_t ws_size,
                              hipStream_t stream) {
  const float* y   = (const float*)d_in[0];
  const float* wgt = (const float*)d_in[1];
  // d_in[2] is step (==2); kernel specialized for step 2.
  float* out = (float*)d_out;

  // workspace: [0, 2MB) weight B-fragments (bf16), [8MB, ~59MB) padded bf16 y
  unsigned short* bpack = (unsigned short*)d_ws;
  unsigned short* ybf   = (unsigned short*)((char*)d_ws + (8u << 20));

  constexpr int PACK_W_TOTAL = NKG * NCOG * 32 * 16;
  pack_weights_bf16<<<(PACK_W_TOTAL + 255) / 256, 256, 0, stream>>>(wgt, bpack);

  const size_t PACK_Y_TOTAL = (size_t)BATCH * NCC * PH * PW * CCHUNK;
  pack_y_bf16<<<(unsigned)((PACK_Y_TOTAL + 255) / 256), 256, 0, stream>>>(y, ybf);

  msc_step2_wmma<<<512 * 3, 256, 0, stream>>>(ybf, bpack, out);
}
